// SelectiveSSMBlock_88021059764654
// MI455X (gfx1250) — matrix operand order repaired and checked by reading, not compiled, not run on hardware
//
#include <hip/hip_runtime.h>
#include <math.h>

// Problem dims (fixed by reference)
#define DIN    256
#define DSTATE 64
#define SEQL   2048
#define BN     64                 // b*n
#define ROWS   (BN * SEQL)        // 131072 flattened rows
#define NCHUNK 16
#define CHL    (SEQL / NCHUNK)    // 128 timesteps per chunk == rows per workgroup

typedef __attribute__((ext_vector_type(16))) __bf16       v16bf;
typedef __attribute__((ext_vector_type(8)))  float        v8f;
typedef __attribute__((ext_vector_type(4)))  unsigned int u32x4;

union FragBF {
    v16bf  v;
    u32x4  u[2];
    __bf16 h[16];
};

__device__ __forceinline__ v8f wmma_bf16(v16bf a, v16bf b, v8f c) {
    // D = A(16x32 bf16) * B(32x16 bf16) + C(16x16 f32)
    return __builtin_amdgcn_wmma_f32_16x16x32_bf16(false, a, false, b, (short)0, c,
                                                   false, false);
}

__device__ __forceinline__ void bf_split(float f, __bf16& hi, __bf16& lo) {
    hi = (__bf16)f;                  // RTNE to bf16
    lo = (__bf16)(f - (float)hi);    // residual; hi+lo ~ fp32
}

// ---------------------------------------------------------------------------
// K0: pre-swizzle Wp (192x256) and Wo (64x64) into WMMA B-fragment layout.
// B frag element (lane, e): K = 32*kstep + 8*(lane/16) + (e&7) + 16*(e/8), N = 16*ntile + lane%16
// Stored [kstep][ntile][lane][16] contiguously -> compute kernels load b128.
// ---------------------------------------------------------------------------
__global__ void k0_preswizzle(const float* __restrict__ Wp, const float* __restrict__ Wo,
                              __bf16* __restrict__ WpH, __bf16* __restrict__ WpL,
                              __bf16* __restrict__ WoH, __bf16* __restrict__ WoL) {
    int idx = blockIdx.x * blockDim.x + threadIdx.x;
    if (idx < 8 * 12 * 32 * 16) {   // Wp fragments: 8 ksteps x 12 ntiles
        int e    = idx & 15;
        int lane = (idx >> 4) & 31;
        int t    = idx >> 9;        // kstep*12 + ntile
        int n    = t % 12;
        int k    = t / 12;
        int K    = k * 32 + 8 * (lane >> 4) + (e & 7) + 16 * (e >> 3);
        int N    = n * 16 + (lane & 15);
        __bf16 h, l;
        bf_split(Wp[N * DIN + K], h, l);
        WpH[idx] = h; WpL[idx] = l;
    }
    if (idx < 2 * 4 * 32 * 16) {    // Wo fragments: 2 ksteps x 4 ntiles
        int e    = idx & 15;
        int lane = (idx >> 4) & 31;
        int n    = (idx >> 9) & 3;
        int k    = idx >> 11;
        int K    = k * 32 + 8 * (lane >> 4) + (e & 7) + 16 * (e >> 3);
        int N    = n * 16 + (lane & 15);
        __bf16 h, l;
        bf_split(Wo[N * DSTATE + K], h, l);
        WoH[idx] = h; WoL[idx] = l;
    }
}

// ---------------------------------------------------------------------------
// F1: fused GEMM1 + pointwise + chunk-local scan.
// Phase A (8 waves, 16 rows each = 128 rows = one scan chunk):
//   p = x @ Wp^T + bp via bf16 hi/lo WMMA;
//   a = sigmoid(p[:,0:64]); u = p[:,64:128]*mean(x,row); c = p[:,128:192];
//   a,u,c -> global (needed by F3); (a,u) also -> LDS for phase B.
// Phase B (64 threads): local scan h_t = a h + u from h=0 over the 128 rows,
//   plus cumulative product of a. Emits per-chunk (prodA, h_end).
// A layout per ISA: lane<16 -> M=lane, e0..7=K0..7, e8..15=K16..23;
//                   lane>=16 -> M=lane-16, e0..7=K8..15, e8..15=K24..31
// ---------------------------------------------------------------------------
__global__ __launch_bounds__(256) void f1_gemm1_scan(
        const float* __restrict__ x, const float* __restrict__ bp,
        const __bf16* __restrict__ WpH, const __bf16* __restrict__ WpL,
        float* __restrict__ Abuf, float* __restrict__ Ubuf, float* __restrict__ Cbuf,
        float* __restrict__ chA, float* __restrict__ chH) {
    // (a,u) interleaved per (t,s): phase-B reads are conflict-free ds_load_b64.
    __shared__ float2 lds_au[CHL * DSTATE];          // 64 KB

    const int wave = threadIdx.x >> 5;
    const int lane = threadIdx.x & 31;
    const int half = lane >> 4;
    const int rows = blockIdx.x * 128 + wave * 16;   // global row base of this wave
    const int mrow = rows + (lane & 15);

    v8f acc[12] = {};
    float sumA = 0.0f;

    for (int k = 0; k < 8; ++k) {
        const float* px = x + (size_t)mrow * DIN + k * 32 + half * 8;
        float4 r0 = ((const float4*)px)[0];
        float4 r1 = ((const float4*)px)[1];
        float4 r2 = ((const float4*)(px + 16))[0];
        float4 r3 = ((const float4*)(px + 16))[1];
        sumA += r0.x + r0.y + r0.z + r0.w + r1.x + r1.y + r1.z + r1.w +
                r2.x + r2.y + r2.z + r2.w + r3.x + r3.y + r3.z + r3.w;

        float rv[16] = { r0.x, r0.y, r0.z, r0.w, r1.x, r1.y, r1.z, r1.w,
                         r2.x, r2.y, r2.z, r2.w, r3.x, r3.y, r3.z, r3.w };
        FragBF ah, al;
#pragma unroll
        for (int e = 0; e < 16; ++e) bf_split(rv[e], ah.h[e], al.h[e]);

#pragma unroll
        for (int n = 0; n < 12; ++n) {
            FragBF bh, bl;
            const u32x4* pb = (const u32x4*)(WpH + ((size_t)(k * 12 + n) * 32 + lane) * 16);
            bh.u[0] = pb[0]; bh.u[1] = pb[1];
            const u32x4* pl = (const u32x4*)(WpL + ((size_t)(k * 12 + n) * 32 + lane) * 16);
            bl.u[0] = pl[0]; bl.u[1] = pl[1];
            acc[n] = wmma_bf16(ah.v, bh.v, acc[n]);
            acc[n] = wmma_bf16(al.v, bh.v, acc[n]);
            acc[n] = wmma_bf16(ah.v, bl.v, acc[n]);
        }
    }

    // Row means: lane r holds half the row-sum of row r%16; combine halves.
    float sfull = sumA + __shfl_xor(sumA, 16, 32);
    float m[8];
#pragma unroll
    for (int i = 0; i < 8; ++i)
        m[i] = __shfl(sfull, i + 8 * half, 32) * (1.0f / 256.0f);

    // C layout: element i -> row_local = i + 8*(lane/16), col = ntile*16 + lane%16
#pragma unroll
    for (int n = 0; n < 12; ++n) {
        float bias = bp[n * 16 + (lane & 15)];
        int   gcol = (n & 3) * 16 + (lane & 15);
        float* outp = (n < 4) ? Abuf : ((n < 8) ? Ubuf : Cbuf);
#pragma unroll
        for (int i = 0; i < 8; ++i) {
            int   rl  = i + 8 * half;            // row within wave strip
            int   tl  = wave * 16 + rl;          // row within chunk (0..127)
            float p   = acc[n][i] + bias;
            float v;
            if (n < 4) {
                v = 1.0f / (1.0f + expf(-p));                    // a
                lds_au[tl * DSTATE + gcol].x = v;
            } else if (n < 8) {
                v = p * m[i];                                    // u
                lds_au[tl * DSTATE + gcol].y = v;
            } else {
                v = p;                                           // c
            }
            outp[(size_t)(rows + rl) * DSTATE + gcol] = v;
        }
    }

    __syncthreads();

    // Phase B: chunk-local scan over 128 timesteps (one thread per state).
    if (threadIdx.x < DSTATE) {
        int s = threadIdx.x;
        float Ap = 1.0f, h = 0.0f;
        for (int t = 0; t < CHL; ++t) {
            float2 au = lds_au[t * DSTATE + s];
            h  = au.x * h + au.y;
            Ap *= au.x;
        }
        size_t c = (size_t)blockIdx.x * DSTATE + s;
        chA[c] = Ap; chH[c] = h;
    }
}

// ---------------------------------------------------------------------------
// K2b: scan over the 16 chunks per (bn, state) -> h at each chunk start.
// ---------------------------------------------------------------------------
__global__ void k2b_interscan(const float* __restrict__ chA, const float* __restrict__ chH,
                              float* __restrict__ hstart) {
    int bn = blockIdx.x, s = threadIdx.x;
    float h = 0.0f;
    for (int ch = 0; ch < NCHUNK; ++ch) {
        size_t idx = (size_t)(bn * NCHUNK + ch) * DSTATE + s;
        hstart[idx] = h;
        h = chA[idx] * h + chH[idx];
    }
}

// ---------------------------------------------------------------------------
// F3: fused scan-finalize + GELU + GEMM2.
// Phase A (64 threads): h_t = a h + u from the true chunk-start h; ot = c*h;
//   g = GELU(ot) split to bf16 hi/lo straight into LDS in A-fragment-friendly
//   row-major (stride 72 bf16 so the 8-element K-runs are 16B aligned).
// Phase B (8 waves): GEMM2 y = g @ Wo^T + bo, A-frags via ds_load_b128.
// ---------------------------------------------------------------------------
__global__ __launch_bounds__(256) void f3_scan_gemm2(
        const float* __restrict__ Abuf, const float* __restrict__ Ubuf,
        const float* __restrict__ Cbuf, const float* __restrict__ hstart,
        const __bf16* __restrict__ WoH, const __bf16* __restrict__ WoL,
        const float* __restrict__ bo, float* __restrict__ y) {
    const int GSTR = 72;                              // bf16 row stride (16B-aligned runs)
    __shared__ __bf16 gh[CHL * GSTR];                 // 18 KB
    __shared__ __bf16 gl[CHL * GSTR];                 // 18 KB

    const int bc = blockIdx.x;
    const size_t row0 = (size_t)bc * 128;

    if (threadIdx.x < DSTATE) {
        int s = threadIdx.x;
        float h = hstart[(size_t)bc * DSTATE + s];
        for (int t = 0; t < CHL; ++t) {
            size_t idx = (row0 + t) * DSTATE + s;
            h = Abuf[idx] * h + Ubuf[idx];
            float ot = Cbuf[idx] * h;
            float g  = 0.5f * ot * (1.0f + erff(ot * 0.70710678118654752f)); // exact GELU
            __bf16 hi, lo;
            bf_split(g, hi, lo);
            gh[t * GSTR + s] = hi;
            gl[t * GSTR + s] = lo;
        }
    }
    __syncthreads();

    const int wave = threadIdx.x >> 5;
    const int lane = threadIdx.x & 31;
    const int half = lane >> 4;
    const int rloc = wave * 16 + (lane & 15);         // row within chunk for A frag

    v8f acc[4] = {};
#pragma unroll
    for (int k = 0; k < 2; ++k) {
        int colbase = k * 32 + half * 8;
        FragBF ah, al;
        ah.u[0] = *(const u32x4*)&gh[rloc * GSTR + colbase];
        ah.u[1] = *(const u32x4*)&gh[rloc * GSTR + colbase + 16];
        al.u[0] = *(const u32x4*)&gl[rloc * GSTR + colbase];
        al.u[1] = *(const u32x4*)&gl[rloc * GSTR + colbase + 16];
#pragma unroll
        for (int n = 0; n < 4; ++n) {
            FragBF bh, bl;
            const u32x4* pb = (const u32x4*)(WoH + ((size_t)(k * 4 + n) * 32 + lane) * 16);
            bh.u[0] = pb[0]; bh.u[1] = pb[1];
            const u32x4* pl = (const u32x4*)(WoL + ((size_t)(k * 4 + n) * 32 + lane) * 16);
            bl.u[0] = pl[0]; bl.u[1] = pl[1];
            acc[n] = wmma_bf16(ah.v, bh.v, acc[n]);
            acc[n] = wmma_bf16(al.v, bh.v, acc[n]);
            acc[n] = wmma_bf16(ah.v, bl.v, acc[n]);
        }
    }
#pragma unroll
    for (int n = 0; n < 4; ++n) {
        float bias = bo[n * 16 + (lane & 15)];
#pragma unroll
        for (int i = 0; i < 8; ++i) {
            int rl = i + 8 * half;
            y[(row0 + wave * 16 + rl) * DSTATE + n * 16 + (lane & 15)] = acc[n][i] + bias;
        }
    }
}

// ---------------------------------------------------------------------------
// Launch. Inputs: x[8,8,2048,256] f32, Wp[192,256], bp[192], Wo[64,64], bo[64].
// Output: y[8,8,2048,64] f32. Workspace need: ~101.6 MB.
// Global traffic: read x 134MB + write a,u,c 100MB + read a,u,c 100MB + write
// y 33.5MB ~= 368MB -> ~16us at 23.3 TB/s.
// ---------------------------------------------------------------------------
extern "C" void kernel_launch(void* const* d_in, const int* in_sizes, int n_in,
                              void* d_out, int out_size, void* d_ws, size_t ws_size,
                              hipStream_t stream) {
    const float* x  = (const float*)d_in[0];
    const float* Wp = (const float*)d_in[1];
    const float* bp = (const float*)d_in[2];
    const float* Wo = (const float*)d_in[3];
    const float* bo = (const float*)d_in[4];
    float*       y  = (float*)d_out;

    char* w = (char*)d_ws;
    const size_t SZ_F = (size_t)ROWS * DSTATE * sizeof(float);       // 33.55 MB
    const size_t SZ_C = (size_t)BN * NCHUNK * DSTATE * sizeof(float);

    float*  Abuf   = (float*)w;            w += SZ_F;
    float*  Ubuf   = (float*)w;            w += SZ_F;
    float*  Cbuf   = (float*)w;            w += SZ_F;
    float*  chA    = (float*)w;            w += SZ_C;
    float*  chH    = (float*)w;            w += SZ_C;
    float*  hstart = (float*)w;            w += SZ_C;
    __bf16* WpH    = (__bf16*)w;           w += (size_t)8 * 12 * 32 * 16 * sizeof(__bf16);
    __bf16* WpL    = (__bf16*)w;           w += (size_t)8 * 12 * 32 * 16 * sizeof(__bf16);
    __bf16* WoH    = (__bf16*)w;           w += (size_t)2 * 4 * 32 * 16 * sizeof(__bf16);
    __bf16* WoL    = (__bf16*)w;

    k0_preswizzle<<<192, 256, 0, stream>>>(Wp, Wo, WpH, WpL, WoH, WoL);
    f1_gemm1_scan<<<ROWS / 128, 256, 0, stream>>>(x, bp, WpH, WpL,
                                                  Abuf, Ubuf, Cbuf, chA, chH);
    k2b_interscan<<<BN, DSTATE, 0, stream>>>(chA, chH, hstart);
    f3_scan_gemm2<<<ROWS / 128, 256, 0, stream>>>(Abuf, Ubuf, Cbuf, hstart,
                                                  WoH, WoL, bo, y);
}
